// TimeTransformer_11716670783627
// MI455X (gfx1250) — matrix operand order repaired and checked by reading, not compiled, not run on hardware
//
#include <hip/hip_runtime.h>
#include <hip/hip_bf16.h>

// Problem constants (match reference)
#define NLAY 3
#define BB 8
#define SS 2
#define EE 130
#define TT 10
#define DD 128
#define HH 8
#define HDIM 16
#define LSEQ (EE*TT)        // 1300
#define NTOK (BB*SS*EE*TT)  // 20800

typedef __attribute__((ext_vector_type(16))) _Float16 v16h;
typedef __attribute__((ext_vector_type(8)))  _Float16 h8;
typedef __attribute__((ext_vector_type(8)))  float    v8f;
typedef __attribute__((ext_vector_type(4)))  unsigned int u32x4;
typedef __attribute__((ext_vector_type(8)))  unsigned int u32x8;

__device__ __forceinline__ int imin(int a, int b){ return a < b ? a : b; }

// ---------------------------------------------------------------------------
// Tensor Data Mover: 2D tile load (rows x 16 fp32, row stride 128 floats)
// from global memory into LDS. Two-SGPR-group descriptor (2D tensor), per
// CDNA5 ISA. Rows beyond tensor_dim1 read back zero,
// which serves as tail padding. Completion tracked with TENSORcnt.
// ---------------------------------------------------------------------------
__device__ __forceinline__ void tdm_load_tile16(unsigned lds_off, const float* gptr,
                                                unsigned rows){
    unsigned long long ga = (unsigned long long)(uintptr_t)gptr;
    u32x4 g0;
    g0[0] = 1u;                                            // count=1, user mode
    g0[1] = lds_off;                                       // lds_addr (bytes)
    g0[2] = (unsigned)(ga & 0xffffffffu);                  // global_addr[31:0]
    g0[3] = (unsigned)((ga >> 32) & 0x01ffffffu) | (2u<<30); // addr[56:32] | type=2
    u32x8 g1;
    g1[0] = (2u << 16);        // workgroup_mask=0, data_size=2 (4 bytes)
    g1[1] = (16u << 16);       // tensor_dim0[15:0]=16 (bits 63:48)
    g1[2] = (rows << 16);      // tensor_dim0 hi=0, tensor_dim1[15:0]=rows
    g1[3] = (16u << 16);       // tensor_dim1 hi=0, tile_dim0=16 (bits 127:112)
    g1[4] = 16u;               // tile_dim1=16, tile_dim2=0
    g1[5] = 128u;              // tensor_dim0_stride = 128 elements
    g1[6] = 0u;                // stride hi, tensor_dim1_stride lo
    g1[7] = 0u;                // tensor_dim1_stride hi
    asm volatile("tensor_load_to_lds %0, %1" :: "s"(g0), "s"(g1) : "memory");
}

// ---------------------------------------------------------------------------
// fp32 -> f16 conversion (weights)
// ---------------------------------------------------------------------------
__global__ void cvt_f16_kernel(const float* __restrict__ in, _Float16* __restrict__ out, int n){
    int i = blockIdx.x*blockDim.x + threadIdx.x;
    if (i < n) out[i] = (_Float16)in[i];
}

// ---------------------------------------------------------------------------
// LayerNorm: one wave (32 lanes) per 128-dim token; f16 output for WMMA A.
// ---------------------------------------------------------------------------
__global__ void ln_kernel(const float* __restrict__ x, const float* __restrict__ g,
                          const float* __restrict__ be, _Float16* __restrict__ outh){
    int token = (blockIdx.x*blockDim.x + threadIdx.x) >> 5;
    int lane  = threadIdx.x & 31;
    if (token >= NTOK) return;
    const float* xp = x + (size_t)token*DD;
    float v[4]; float s = 0.f;
    #pragma unroll
    for (int i=0;i<4;i++){ v[i] = xp[lane + 32*i]; s += v[i]; }
    #pragma unroll
    for (int off=16; off>0; off>>=1) s += __shfl_xor(s, off, 32);
    float mean = s * (1.f/DD);
    float var = 0.f;
    #pragma unroll
    for (int i=0;i<4;i++){ float d = v[i]-mean; var += d*d; }
    #pragma unroll
    for (int off=16; off>0; off>>=1) var += __shfl_xor(var, off, 32);
    float inv = rsqrtf(var*(1.f/DD) + 1e-5f);
    _Float16* op = outh + (size_t)token*DD;
    #pragma unroll
    for (int i=0;i<4;i++){
        int c = lane + 32*i;
        op[c] = (_Float16)((v[i]-mean)*inv*g[c] + be[c]);
    }
}

// ---------------------------------------------------------------------------
// WMMA GEMM: out[N x Dout] = A[N x K](f16) @ W[Dout x K]^T(f16) + bias
// One wave per 16x16 tile; 8 waves per block cover 128 output columns.
// MODE 0: fp32 out. MODE 1: fp32 out + residual. MODE 2: exact-GELU -> f16.
// Fragment layouts follow CDNA5 ISA §7.12.2 (wave32).
// ---------------------------------------------------------------------------
template<int MODE>
__global__ __launch_bounds__(256) void gemm_kernel(
        const _Float16* __restrict__ A, const _Float16* __restrict__ W,
        const float* __restrict__ bias, const float* __restrict__ res,
        float* __restrict__ outf, _Float16* __restrict__ outh, int K, int Dout){
    int wv     = threadIdx.x >> 5;
    int lane   = threadIdx.x & 31;
    int lane15 = lane & 15;
    int hig    = lane >> 4;
    int mb = blockIdx.x * 16;
    int nb = (blockIdx.y*8 + wv) * 16;

    v8f acc = {};
    const _Float16* arow = A + (size_t)(mb + lane15)*K;   // A: row M=lane%16
    const _Float16* brow = W + (size_t)(nb + lane15)*K;   // B col N=lane%16 = W row

    for (int kb = 0; kb < K; kb += 32) {
        // A 16x32 f16: halfs 0..7 -> K = kb + hig*8 + i ; halfs 8..15 -> +16
        h8 a0 = *(const h8*)(arow + kb + hig*8);
        h8 a1 = *(const h8*)(arow + kb + hig*8 + 16);
        // B 32x16 f16: halfs 0..7 -> K = kb + hig*8 + i ; halfs 8..15 -> +16
        h8 b0 = *(const h8*)(brow + kb + hig*8);
        h8 b1 = *(const h8*)(brow + kb + hig*8 + 16);
        v16h a, b;
        #pragma unroll
        for (int i=0;i<8;i++){ a[i]=a0[i]; a[8+i]=a1[i]; b[i]=b0[i]; b[8+i]=b1[i]; }
        acc = __builtin_amdgcn_wmma_f32_16x16x32_f16(false, a, false, b,
                                                     (short)0, acc, false, false);
    }

    int n = nb + lane15;
    float bv = bias[n];
    #pragma unroll
    for (int vv=0; vv<8; vv++){
        int mrow = mb + vv + hig*8;                  // C/D: VGPR v -> M = v + hi*8
        size_t idx = (size_t)mrow*Dout + n;
        float val = acc[vv] + bv;
        if (MODE == 1) val += res[idx];
        if (MODE == 2) {
            float gl = 0.5f*val*(1.f + erff(val*0.70710678118f));
            outh[idx] = (_Float16)gl;
        } else {
            outf[idx] = val;
        }
    }
}

// ---------------------------------------------------------------------------
// RoPE (fp32, in-place on q and k); matches reference theta = 1e4^floor(-a/HD/2)
// ---------------------------------------------------------------------------
__global__ void rope_kernel(float* __restrict__ q, float* __restrict__ k){
    int i = blockIdx.x*blockDim.x + threadIdx.x;
    if (i >= NTOK*HH) return;
    int token = i >> 3, h = i & 7;
    int t = token % TT;
    float* qp = q + (size_t)token*DD + h*HDIM;
    float* kp = k + (size_t)token*DD + h*HDIM;
    #pragma unroll
    for (int j=0;j<4;j++){
        float a = 2.f*(float)j;
        float theta = __powf(10000.f, floorf(-a/(float)HDIM*0.5f));
        float pos = (float)t * theta;
        float c = __cosf(pos), s = __sinf(pos);
        float ql = qp[j], qr = qp[4+j];
        qp[j]   = ql*c + qr*s;
        qp[4+j] = qr*c - ql*s;
        float kl = kp[j], kr = kp[4+j];
        kp[j]   = kl*c + kr*s;
        kp[4+j] = kr*c - kl*s;
    }
}

// ---------------------------------------------------------------------------
// Flash-style attention: one wave per 16 query rows per (batch, head, pos/neg).
// K/V tiles are staged into LDS by the Tensor Data Mover (double-buffered,
// TENSORcnt-tracked), so the inner loop overlaps DMA with WMMA + softmax.
// Score WMMA: K=HD=16 real (zero-padded to 32). Online softmax via width-16
// shuffles. P converted C-layout -> A-layout through LDS. Causal mask applies
// on t = index % T only (per reference broadcast).
// ---------------------------------------------------------------------------
__global__ __launch_bounds__(32) void attn_kernel(const float* __restrict__ q,
        const float* __restrict__ k, const float* __restrict__ v,
        _Float16* __restrict__ oh){
    __shared__ float smK[2][16*16];   // [buf][key_local*16 + hd]
    __shared__ float smV[2][16*16];
    __shared__ float smP[16][24];
    int lane   = threadIdx.x;
    int lane15 = lane & 15;
    int hig    = lane >> 4;
    int c = blockIdx.y;              // c = b*16 + h*2 + m
    int b = c >> 4, h = (c >> 1) & 7, m = c & 1;
    const int qbase  = (b*SS + m)*LSEQ;   // queries: s = m
    const int kvbase = (b*SS)*LSEQ;       // keys/values: s = 0
    int q0 = blockIdx.x*16;

    const unsigned kOff = (unsigned)(uintptr_t)&smK[0][0];
    const unsigned vOff = (unsigned)(uintptr_t)&smV[0][0];

    // Q fragment: A 16x32, M=lane%16, halfs 0..7 -> hd = hig*8 + i, rest zero.
    int qrow = q0 + lane15;
    const float* qp = q + (size_t)(qbase + imin(qrow, LSEQ-1))*DD + h*HDIM + hig*8;
    v16h aq;
    #pragma unroll
    for (int i=0;i<8;i++){ aq[i] = (_Float16)(qp[i]*0.25f); aq[8+i] = (_Float16)0.f; }

    float mrow[8], lrow[8];
    v8f acc = {};
    #pragma unroll
    for (int vv=0; vv<8; vv++){ mrow[vv] = -1e30f; lrow[vv] = 0.f; }

    const int nkc = (LSEQ + 15)/16;   // 82 key chunks

    // Prefetch chunk 0 into buffer 0 via TDM.
    {
        const float* kt = k + (size_t)(kvbase)*DD + h*HDIM;
        const float* vt = v + (size_t)(kvbase)*DD + h*HDIM;
        tdm_load_tile16(kOff, kt, 16u);
        tdm_load_tile16(vOff, vt, 16u);
    }

    for (int kc = 0; kc < nkc; kc++){
        int cur = kc & 1;
        // Issue next chunk's TDM loads into the other buffer, then wait for
        // the current pair (in-order per-wave TENSORcnt: <=2 -> current done).
        if (kc + 1 < nkc){
            int nxt = cur ^ 1;
            unsigned rows = (unsigned)imin(16, LSEQ - (kc+1)*16);
            const float* kt = k + (size_t)(kvbase + (kc+1)*16)*DD + h*HDIM;
            const float* vt = v + (size_t)(kvbase + (kc+1)*16)*DD + h*HDIM;
            tdm_load_tile16(kOff + (unsigned)nxt*16*16*4, kt, rows);
            tdm_load_tile16(vOff + (unsigned)nxt*16*16*4, vt, rows);
            __builtin_amdgcn_s_wait_tensorcnt(2);
        } else {
            __builtin_amdgcn_s_wait_tensorcnt(0);
        }
        asm volatile("" ::: "memory");

        int key = kc*16 + lane15;
        // K^T fragment: B 32x16, N=key(lane%16), halfs 0..7 -> hd = hig*8 + i, rest zero
        v16h bk;
        {
            const float* kp = &smK[cur][lane15*16 + hig*8];
            #pragma unroll
            for (int i=0;i<8;i++){ bk[i] = (_Float16)kp[i]; bk[8+i] = (_Float16)0.f; }
        }
        v8f zero = {};
        v8f s = __builtin_amdgcn_wmma_f32_16x16x32_f16(false, aq, false, bk,
                                                       (short)0, zero, false, false);
        bool keyok = key < LSEQ;
        int tk = key % TT;
        float pv[8];
        #pragma unroll
        for (int vv=0; vv<8; vv++){
            int row = q0 + vv + hig*8;
            int tq = row % TT;
            float sv = s[vv];
            if (!keyok || tk > tq) sv = -1e30f;     // causal over t only
            float mx = sv;
            #pragma unroll
            for (int off=8; off>0; off>>=1) mx = fmaxf(mx, __shfl_xor(mx, off, 16));
            float nm = fmaxf(mrow[vv], mx);
            float p  = __expf(sv - nm);
            float sum = p;
            #pragma unroll
            for (int off=8; off>0; off>>=1) sum += __shfl_xor(sum, off, 16);
            float scale = __expf(mrow[vv] - nm);
            lrow[vv] = lrow[vv]*scale + sum;
            mrow[vv] = nm;
            acc[vv] *= scale;
            pv[vv]   = p;
        }
        // C-layout (row=v+hi*8, col=lane%16) -> LDS -> A-layout (row=lane%16)
        #pragma unroll
        for (int vv=0; vv<8; vv++) smP[vv + hig*8][lane15] = pv[vv];
        __syncthreads();
        v16h ap;
        #pragma unroll
        for (int i=0;i<8;i++){ ap[i] = (_Float16)smP[lane15][i + hig*8];
                               ap[8+i] = (_Float16)0.f; }
        // V fragment: B 32x16, K=key_local (0..15 real), N=hd=lane%16
        v16h bvf;
        {
            #pragma unroll
            for (int i=0;i<8;i++){ bvf[i] = (_Float16)smV[cur][(i + hig*8)*16 + lane15];
                                   bvf[8+i] = (_Float16)0.f; }
        }
        acc = __builtin_amdgcn_wmma_f32_16x16x32_f16(false, ap, false, bvf,
                                                     (short)0, acc, false, false);
        // Drain LDS reads before this buffer is overwritten by a later TDM.
        __syncthreads();
    }
    #pragma unroll
    for (int vv=0; vv<8; vv++){
        int row = q0 + vv + hig*8;
        if (row < LSEQ){
            float o = acc[vv] / lrow[vv];
            oh[(size_t)(qbase + row)*DD + h*HDIM + lane15] = (_Float16)o;
        }
    }
}

// ---------------------------------------------------------------------------
// Final projection: out[n] = dot(x[n,:], Wf) + bf
// ---------------------------------------------------------------------------
__global__ void final_kernel(const float* __restrict__ x, const float* __restrict__ Wf,
                             const float* __restrict__ bf, float* __restrict__ out){
    int token = (blockIdx.x*blockDim.x + threadIdx.x) >> 5;
    int lane  = threadIdx.x & 31;
    if (token >= NTOK) return;
    const float* xp = x + (size_t)token*DD;
    float s = 0.f;
    #pragma unroll
    for (int i=0;i<4;i++) s += xp[lane + 32*i]*Wf[lane + 32*i];
    #pragma unroll
    for (int off=16; off>0; off>>=1) s += __shfl_xor(s, off, 32);
    if (lane == 0) out[token] = s + bf[0];
}

// ---------------------------------------------------------------------------
extern "C" void kernel_launch(void* const* d_in, const int* in_sizes, int n_in,
                              void* d_out, int out_size, void* d_ws, size_t ws_size,
                              hipStream_t stream){
    (void)in_sizes; (void)n_in; (void)out_size; (void)ws_size;
    const float* x   = (const float*)d_in[0];
    const float* Wq  = (const float*)d_in[1];
    const float* bq  = (const float*)d_in[2];
    const float* Wk  = (const float*)d_in[3];
    const float* bk  = (const float*)d_in[4];
    const float* Wv  = (const float*)d_in[5];
    const float* bv  = (const float*)d_in[6];
    const float* Wo  = (const float*)d_in[7];
    const float* bo  = (const float*)d_in[8];
    const float* W1  = (const float*)d_in[9];
    const float* b1  = (const float*)d_in[10];
    const float* W2  = (const float*)d_in[11];
    const float* b2  = (const float*)d_in[12];
    const float* g1  = (const float*)d_in[13];
    const float* be1 = (const float*)d_in[14];
    const float* g2  = (const float*)d_in[15];
    const float* be2 = (const float*)d_in[16];
    const float* Wf  = (const float*)d_in[17];
    const float* bf  = (const float*)d_in[18];
    float* out = (float*)d_out;

    // Workspace carve-out (256B aligned)
    char* p = (char*)d_ws;
    auto carve = [&](size_t bytes)->char*{
        char* r = p; p += (bytes + 255) & ~(size_t)255; return r;
    };
    float*    xcur = (float*)   carve((size_t)NTOK*DD*4);
    float*    qf   = (float*)   carve((size_t)NTOK*DD*4);
    float*    kf   = (float*)   carve((size_t)NTOK*DD*4);
    float*    vf   = (float*)   carve((size_t)NTOK*DD*4);
    _Float16* x2h  = (_Float16*)carve((size_t)NTOK*DD*2);
    _Float16* ohb  = (_Float16*)carve((size_t)NTOK*DD*2);
    _Float16* h1h  = (_Float16*)carve((size_t)NTOK*4*DD*2);
    _Float16* wqh  = (_Float16*)carve((size_t)DD*DD*2);
    _Float16* wkh  = (_Float16*)carve((size_t)DD*DD*2);
    _Float16* wvh  = (_Float16*)carve((size_t)DD*DD*2);
    _Float16* woh  = (_Float16*)carve((size_t)DD*DD*2);
    _Float16* w1h  = (_Float16*)carve((size_t)4*DD*DD*2);
    _Float16* w2h  = (_Float16*)carve((size_t)4*DD*DD*2);

    hipMemcpyAsync(xcur, x, (size_t)NTOK*DD*4, hipMemcpyDeviceToDevice, stream);

    const int nDD = DD*DD, n4DD = 4*DD*DD;
    dim3 gemmBlk(256);
    dim3 gemmGrid1(NTOK/16, 1);     // Dout = 128
    dim3 gemmGrid4(NTOK/16, 4);     // Dout = 512
    dim3 attnGrid((LSEQ+15)/16, BB*HH*SS);

    for (int i=0;i<NLAY;i++){
        const float* Wqi = Wq + (size_t)i*nDD;
        const float* Wki = Wk + (size_t)i*nDD;
        const float* Wvi = Wv + (size_t)i*nDD;
        const float* Woi = Wo + (size_t)i*nDD;
        const float* W1i = W1 + (size_t)i*n4DD;
        const float* W2i = W2 + (size_t)i*n4DD;

        cvt_f16_kernel<<<(nDD+255)/256, 256, 0, stream>>>(Wqi, wqh, nDD);
        cvt_f16_kernel<<<(nDD+255)/256, 256, 0, stream>>>(Wki, wkh, nDD);
        cvt_f16_kernel<<<(nDD+255)/256, 256, 0, stream>>>(Wvi, wvh, nDD);
        cvt_f16_kernel<<<(nDD+255)/256, 256, 0, stream>>>(Woi, woh, nDD);
        cvt_f16_kernel<<<(n4DD+255)/256, 256, 0, stream>>>(W1i, w1h, n4DD);
        cvt_f16_kernel<<<(n4DD+255)/256, 256, 0, stream>>>(W2i, w2h, n4DD);

        // LN1 -> f16 activations
        ln_kernel<<<NTOK/4, 128, 0, stream>>>(xcur, g1 + i*DD, be1 + i*DD, x2h);
        // Q/K/V projections (WMMA)
        gemm_kernel<0><<<gemmGrid1, gemmBlk, 0, stream>>>(x2h, wqh, bq + i*DD,
            nullptr, qf, nullptr, DD, DD);
        gemm_kernel<0><<<gemmGrid1, gemmBlk, 0, stream>>>(x2h, wkh, bk + i*DD,
            nullptr, kf, nullptr, DD, DD);
        gemm_kernel<0><<<gemmGrid1, gemmBlk, 0, stream>>>(x2h, wvh, bv + i*DD,
            nullptr, vf, nullptr, DD, DD);
        // RoPE on q/k
        rope_kernel<<<(NTOK*HH+255)/256, 256, 0, stream>>>(qf, kf);
        // Attention (pos: s=0 queries, neg: s=1 queries; kv always s=0)
        attn_kernel<<<attnGrid, 32, 0, stream>>>(qf, kf, vf, ohb);
        // O-projection + residual
        gemm_kernel<1><<<gemmGrid1, gemmBlk, 0, stream>>>(ohb, woh, bo + i*DD,
            xcur, xcur, nullptr, DD, DD);
        // LN2 -> f16
        ln_kernel<<<NTOK/4, 128, 0, stream>>>(xcur, g2 + i*DD, be2 + i*DD, x2h);
        // FFN1 with exact GELU -> f16
        gemm_kernel<2><<<gemmGrid4, gemmBlk, 0, stream>>>(x2h, w1h, b1 + i*4*DD,
            nullptr, nullptr, h1h, DD, 4*DD);
        // FFN2 + residual
        gemm_kernel<1><<<gemmGrid1, gemmBlk, 0, stream>>>(h1h, w2h, b2 + i*DD,
            xcur, xcur, nullptr, 4*DD, DD);
    }
    final_kernel<<<NTOK/4, 128, 0, stream>>>(xcur, Wf, bf, out);
}
